// NADE_6622839570841
// MI455X (gfx1250) — matrix operand order repaired and checked by reading, not compiled, not run on hardware
//
#include <hip/hip_runtime.h>

typedef __attribute__((ext_vector_type(16))) _Float16 v16h;
typedef __attribute__((ext_vector_type(8)))  float    v8f;

#define DD 4096      // input dim (columns i)
#define HH 4096      // hidden dim (rows h)
#define NSTRIP 256   // DD / 16

// ---------------------------------------------------------------------------
// Kernel 1: Base[h][t] = c[h] + sum_{j < 16t} w[h,j] * x[j]
// One 256-thread block per row h. Thread t sums its 16-column strip, then a
// Kogge-Stone scan over the 256 strip sums in LDS gives the exclusive prefix.
// Reads w (64 MB) once; writes 4 MB. Pure streaming, memory-bound.
// ---------------------------------------------------------------------------
__global__ void nade_base_kernel(const float* __restrict__ w,
                                 const float* __restrict__ x,
                                 const float* __restrict__ c,
                                 float* __restrict__ base) {
    __shared__ float s[NSTRIP];
    const int t = threadIdx.x;          // strip index
    const int h = blockIdx.x;           // row
    const float* wr = w + (size_t)h * DD + t * 16;
    const float* xr = x + t * 16;
    float sum = 0.f;
#pragma unroll
    for (int k = 0; k < 16; ++k) sum += wr[k] * xr[k];
    s[t] = sum;
    __syncthreads();
    // inclusive Kogge-Stone scan over 256 strip sums
    for (int off = 1; off < NSTRIP; off <<= 1) {
        float vv = (t >= off) ? s[t - off] : 0.f;
        __syncthreads();
        s[t] += vv;
        __syncthreads();
    }
    float excl = (t == 0) ? 0.f : s[t - 1];
    base[(size_t)h * NSTRIP + t] = c[h] + excl;
}

// ---------------------------------------------------------------------------
// Kernel 2: one block (4 waves) per 16-column strip t.
// Per wave, per 32-row chunk: finish the exclusive scan locally (16 steps),
// sigmoid -> f16 tile in LDS, gather WMMA A (v[i,h] 16x32) and B (sig 32x16)
// fragments in contiguous k order (k = 16*(lane/16) + s), accumulate C with
// v_wmma_f32_16x16x32_f16. Only the diagonal of C is used:
//   logits[16t+i] = C[i][i] + b[16t+i];  out = sigmoid(logits).
// ---------------------------------------------------------------------------
__global__ void nade_main_kernel(const float* __restrict__ w,
                                 const float* __restrict__ x,
                                 const float* __restrict__ b,
                                 const float* __restrict__ v,
                                 const float* __restrict__ base,
                                 float* __restrict__ out) {
    __shared__ float    xs[16];
    __shared__ _Float16 sigT[4][32][16];   // per-wave sigmoid tile [h][i]
    __shared__ float    cred[4][16][16];   // per-wave C tiles for reduction

    const int t    = blockIdx.x;           // strip: columns i in [16t, 16t+16)
    const int lane = threadIdx.x & 31;
    const int wave = threadIdx.x >> 5;

    if (threadIdx.x < 16) xs[threadIdx.x] = x[t * 16 + threadIdx.x];
    __syncthreads();

    const int n  = lane & 15;              // N (and M) index within tile
    const int kb = (lane < 16) ? 0 : 16;   // B-fragment K base for this lane half
    const int iglob = t * 16 + n;          // global column handled by A-fragment row
    const float* vrow_base = v + (size_t)iglob * HH;

    v8f acc = {};                          // f32 16x16 accumulator (8 VGPRs)

    for (int cchunk = wave; cchunk < HH / 32; cchunk += 4) {
        const int h0 = cchunk * 32;
        const int h  = h0 + lane;

        // prefetch next chunk's v / w segments into caches (global_prefetch_b8)
        __builtin_prefetch(vrow_base + h0 + 128, 0, 1);
        __builtin_prefetch(w + (size_t)(h + 128) * DD + t * 16, 0, 1);

        // exclusive local scan + sigmoid for row h over the 16 strip columns
        float a = base[(size_t)h * NSTRIP + t];
        const float* wr = w + (size_t)h * DD + t * 16;
#pragma unroll
        for (int k = 0; k < 16; ++k) {
            float s = 1.0f / (1.0f + __expf(-a));   // sigmoid(A[h, 16t+k])
            sigT[wave][lane][k] = (_Float16)s;
            a += wr[k] * xs[k];                     // advance exclusive prefix
        }
        __syncthreads();   // make f16 tile visible across lanes (ds wait)

        // B fragment: 32x16 f16. Lanes 0-15 hold K=0..15 (N=lane),
        // lanes 16-31 hold K=16..31 (N=lane-16); VGPR r packs K=2r,2r+1.
        v16h bf;
#pragma unroll
        for (int j = 0; j < 16; ++j) bf[j] = sigT[wave][kb + j][n];

        // A fragment: v tile 16x32 f16, M = lane%16 = i-within-strip.
        // lanes 0-15: K = 0..15; lanes 16-31: K = 16..31.
        v16h af;
#pragma unroll
        for (int j = 0; j < 16; ++j) {
            int koff = j;
            koff += (lane < 16) ? 0 : 16;
            af[j] = (_Float16)vrow_base[h0 + koff];
        }

        acc = __builtin_amdgcn_wmma_f32_16x16x32_f16(
            /*neg_a=*/false, af, /*neg_b=*/false, bf,
            /*c_mod=*/(short)0, acc, /*reuse_a=*/false, /*reuse_b=*/false);
        __syncthreads();
    }

    // C layout: lanes 0-15: M=r, N=lane; lanes 16-31: M=8+r, N=lane-16
#pragma unroll
    for (int r = 0; r < 8; ++r) {
        int m = (lane < 16) ? r : (8 + r);
        cred[wave][m][n] = acc[r];
    }
    __syncthreads();

    if (threadIdx.x < 16) {
        int i = threadIdx.x;
        float d = cred[0][i][i] + cred[1][i][i] + cred[2][i][i] + cred[3][i][i];
        float logit = d + b[t * 16 + i];
        out[t * 16 + i] = 1.0f / (1.0f + __expf(-logit));
    }
}

// ---------------------------------------------------------------------------
// Inputs (setup_inputs order): x[4096], w[4096*4096], b[4096], v[4096*4096],
// c[4096]; all float32. Output: float32 p_x[4096]. Workspace: Base = 4 MB.
// ---------------------------------------------------------------------------
extern "C" void kernel_launch(void* const* d_in, const int* in_sizes, int n_in,
                              void* d_out, int out_size, void* d_ws, size_t ws_size,
                              hipStream_t stream) {
    (void)in_sizes; (void)n_in; (void)out_size; (void)ws_size;
    const float* x = (const float*)d_in[0];
    const float* w = (const float*)d_in[1];
    const float* b = (const float*)d_in[2];
    const float* v = (const float*)d_in[3];
    const float* c = (const float*)d_in[4];
    float* out  = (float*)d_out;
    float* base = (float*)d_ws;   // HH * NSTRIP floats = 4 MB

    nade_base_kernel<<<HH, NSTRIP, 0, stream>>>(w, x, c, base);
    nade_main_kernel<<<NSTRIP, 128, 0, stream>>>(w, x, b, v, base, out);
}
